// DeepseekMLA_51281909514560
// MI455X (gfx1250) — compile-verified, matrix operand order checked by reading, not run on hardware
//
#include <hip/hip_runtime.h>

// ---------------------------------------------------------------------------
// DeepSeek MLA prefill for MI455X (gfx1250, wave32, WMMA bf16).
// All GEMMs + attention matmuls go through v_wmma_f32_16x16x32_bf16.
// Attention stages K/V tiles in LDS, shared by 4 waves (64 query rows/block).
// ---------------------------------------------------------------------------

typedef __attribute__((ext_vector_type(16))) __bf16 v16bf;
typedef __attribute__((ext_vector_type(8)))  __bf16 v8bf;
typedef __attribute__((ext_vector_type(8)))  float  v8f;

#define T_TOK 2048
#define NHEAD 16
#define DNOPE 128
#define DROPE 64
#define DVAL  128
#define QLAT  1536
#define KVLAT 512
#define HID   5120
#define ATT_SCALE 0.07216878364870323f   // (128+64)^-0.5
#define RMS_EPS 1e-6f
#define ROPE_THETA 10000.0f

// ---------------------------------------------------------------------------
// Fragment loader for 16-bit WMMA operands (A: [M,K] row-major, B: W^T [N,K]
// row-major). ISA layout: lane l -> row (l&15); K chunks at (l>>4)*8 and
// 16+(l>>4)*8, each 8 contiguous bf16 => two b128 loads (global or ds).
// ---------------------------------------------------------------------------
__device__ __forceinline__ v16bf load_frag16(const __bf16* base, int ld,
                                             int row0, int k0) {
    int lane = threadIdx.x & 31;
    int r  = row0 + (lane & 15);
    int ks = k0 + ((lane >> 4) << 3);
    const __bf16* p = base + (size_t)r * (size_t)ld + ks;
    v8bf lo = *(const v8bf*)(p);
    v8bf hi = *(const v8bf*)(p + 16);
    v16bf f;
#pragma unroll
    for (int i = 0; i < 8; ++i) { f[i] = lo[i]; f[8 + i] = hi[i]; }
    return f;
}

// ---------------------------------------------------------------------------
// Generic bf16 GEMM: C[M,N](f32) = A[M,K] * B[K,N], B given as BT[N,K].
// 128 threads = 4 waves; block tile 128x128; wave tile 64x64 (16 WMMAs/kstep).
// M, N multiples of 128; K multiple of 32.
// ---------------------------------------------------------------------------
__global__ __launch_bounds__(128) void gemm_bf16_wmma(const __bf16* __restrict__ A,
                                                      const __bf16* __restrict__ BT,
                                                      float* __restrict__ C,
                                                      int M, int N, int K) {
    const int wave = threadIdx.x >> 5;
    const int lane = threadIdx.x & 31;
    const int m0 = blockIdx.y * 128 + (wave >> 1) * 64;
    const int n0 = blockIdx.x * 128 + (wave & 1) * 64;

    v8f acc[4][4];
#pragma unroll
    for (int i = 0; i < 4; ++i)
#pragma unroll
        for (int j = 0; j < 4; ++j) acc[i][j] = 0.f;

    for (int k0 = 0; k0 < K; k0 += 32) {
        v16bf a[4], b[4];
#pragma unroll
        for (int i = 0; i < 4; ++i) a[i] = load_frag16(A,  K, m0 + 16 * i, k0);
#pragma unroll
        for (int j = 0; j < 4; ++j) b[j] = load_frag16(BT, K, n0 + 16 * j, k0);
#pragma unroll
        for (int i = 0; i < 4; ++i)
#pragma unroll
            for (int j = 0; j < 4; ++j)
                acc[i][j] = __builtin_amdgcn_wmma_f32_16x16x32_bf16(
                    false, a[i], false, b[j], (short)0, acc[i][j], false, false);
    }

    // C/D layout: element (M = r + 8*(lane>>4), N = lane&15) in VGPR r.
    const int ro = (lane >> 4) << 3;
    const int co = lane & 15;
#pragma unroll
    for (int i = 0; i < 4; ++i)
#pragma unroll
        for (int j = 0; j < 4; ++j)
#pragma unroll
            for (int r = 0; r < 8; ++r)
                C[(size_t)(m0 + 16 * i + r + ro) * (size_t)N + (n0 + 16 * j + co)] =
                    acc[i][j][r];
}

// ---------------------------------------------------------------------------
// fp32 -> bf16 elementwise
// ---------------------------------------------------------------------------
__global__ void conv_bf16(const float* __restrict__ X, __bf16* __restrict__ Y,
                          size_t n) {
    size_t i = (size_t)blockIdx.x * blockDim.x + threadIdx.x;
    if (i < n) Y[i] = (__bf16)X[i];
}

// ---------------------------------------------------------------------------
// Transpose + convert: W[K,N] f32 -> WT[Npad,K] bf16 (zero rows for n>=N).
// ---------------------------------------------------------------------------
__global__ __launch_bounds__(256) void transpose_conv(const float* __restrict__ W,
                                                      __bf16* __restrict__ WT,
                                                      int K, int N, int Npad) {
    __shared__ float tile[32][33];
    const int n0 = blockIdx.x * 32, k0 = blockIdx.y * 32;
    const int tx = threadIdx.x & 31, ty = threadIdx.x >> 5;
#pragma unroll
    for (int i = 0; i < 32; i += 8) {
        int k = k0 + ty + i, n = n0 + tx;
        tile[ty + i][tx] = (k < K && n < N) ? W[(size_t)k * N + n] : 0.f;
    }
    __syncthreads();
#pragma unroll
    for (int i = 0; i < 32; i += 8) {
        int n = n0 + ty + i, k = k0 + tx;
        if (n < Npad && k < K) WT[(size_t)n * K + k] = (__bf16)tile[tx][ty + i];
    }
}

// ---------------------------------------------------------------------------
// RMSNorm row kernel: Y[row,0:width](bf16) = norm(X[row,0:width]) * w
// ---------------------------------------------------------------------------
__global__ __launch_bounds__(256) void rmsnorm_bf16(const float* __restrict__ X,
                                                    const float* __restrict__ w,
                                                    __bf16* __restrict__ Y,
                                                    int width) {
    __shared__ float red[256];
    const int row = blockIdx.x;
    const float* x = X + (size_t)row * width;
    float ss = 0.f;
    for (int c = threadIdx.x; c < width; c += 256) { float v = x[c]; ss += v * v; }
    red[threadIdx.x] = ss;
    __syncthreads();
    for (int s = 128; s > 0; s >>= 1) {
        if ((int)threadIdx.x < s) red[threadIdx.x] += red[threadIdx.x + s];
        __syncthreads();
    }
    const float inv = rsqrtf(red[0] / (float)width + RMS_EPS);
    for (int c = threadIdx.x; c < width; c += 256)
        Y[(size_t)row * width + c] = (__bf16)(x[c] * inv * w[c]);
}

// ---------------------------------------------------------------------------
// kv prep: rmsnorm(kv[:, :512]) -> CKV bf16; rope(kv[:, 512:576]) -> KPE bf16.
// KV has ld = 640 (padded N of the wkv_a GEMM).
// ---------------------------------------------------------------------------
__global__ __launch_bounds__(256) void prep_kv(const float* __restrict__ KV,
                                               const float* __restrict__ w,
                                               __bf16* __restrict__ CKV,
                                               __bf16* __restrict__ KPE) {
    __shared__ float red[256];
    const int t = blockIdx.x;
    const float* x = KV + (size_t)t * 640;
    float ss = 0.f;
    for (int c = threadIdx.x; c < KVLAT; c += 256) { float v = x[c]; ss += v * v; }
    red[threadIdx.x] = ss;
    __syncthreads();
    for (int s = 128; s > 0; s >>= 1) {
        if ((int)threadIdx.x < s) red[threadIdx.x] += red[threadIdx.x + s];
        __syncthreads();
    }
    const float inv = rsqrtf(red[0] / (float)KVLAT + RMS_EPS);
    for (int c = threadIdx.x; c < KVLAT; c += 256)
        CKV[(size_t)t * KVLAT + c] = (__bf16)(x[c] * inv * w[c]);

    if (threadIdx.x < DROPE / 2) {
        int i = threadIdx.x;
        float x1 = x[KVLAT + 2 * i], x2 = x[KVLAT + 2 * i + 1];
        float fr = __powf(ROPE_THETA, -((float)(2 * i) / (float)DROPE));
        float ang = (float)t * fr, sn, cs;
        __sincosf(ang, &sn, &cs);
        KPE[(size_t)t * DROPE + 2 * i]     = (__bf16)(x1 * cs - x2 * sn);
        KPE[(size_t)t * DROPE + 2 * i + 1] = (__bf16)(x2 * cs + x1 * sn);
    }
}

// ---------------------------------------------------------------------------
// Build Qb[h][t][192] bf16 from q f32 [t][h][192]; rope applied to last 64.
// ---------------------------------------------------------------------------
__global__ void build_q(const float* __restrict__ Q, __bf16* __restrict__ Qb) {
    size_t idx = (size_t)blockIdx.x * blockDim.x + threadIdx.x;
    const size_t total = (size_t)T_TOK * NHEAD * 96;
    if (idx >= total) return;
    int j  = (int)(idx % 96);
    int th = (int)(idx / 96);
    int h  = th % NHEAD;
    int t  = th / NHEAD;
    const float* q = Q + (size_t)t * (NHEAD * 192) + (size_t)h * 192;
    __bf16* o = Qb + ((size_t)h * T_TOK + t) * 192;
    if (j < 64) {
        o[2 * j]     = (__bf16)q[2 * j];
        o[2 * j + 1] = (__bf16)q[2 * j + 1];
    } else {
        int i = j - 64;
        float x1 = q[DNOPE + 2 * i], x2 = q[DNOPE + 2 * i + 1];
        float fr = __powf(ROPE_THETA, -((float)(2 * i) / (float)DROPE));
        float ang = (float)t * fr, sn, cs;
        __sincosf(ang, &sn, &cs);
        o[DNOPE + 2 * i]     = (__bf16)(x1 * cs - x2 * sn);
        o[DNOPE + 2 * i + 1] = (__bf16)(x2 * cs + x1 * sn);
    }
}

// ---------------------------------------------------------------------------
// Build Kb[h][t][192] (nope + shared roped k_pe) and Vt[h][dv][t] (transposed
// V for the PV matmul B-operand) from kvb f32 [t][h][256].
// ---------------------------------------------------------------------------
__global__ void build_kv(const float* __restrict__ KVB,
                         const __bf16* __restrict__ KPE,
                         __bf16* __restrict__ Kb, __bf16* __restrict__ Vt) {
    size_t idx = (size_t)blockIdx.x * blockDim.x + threadIdx.x;
    const size_t total = (size_t)NHEAD * T_TOK * 320;
    if (idx >= total) return;
    int d  = (int)(idx % 320);
    int th = (int)(idx / 320);
    int t  = th % T_TOK;
    int h  = th / T_TOK;
    if (d < DNOPE) {
        Kb[((size_t)h * T_TOK + t) * 192 + d] =
            (__bf16)KVB[(size_t)t * (NHEAD * 256) + (size_t)h * 256 + d];
    } else if (d < 192) {
        Kb[((size_t)h * T_TOK + t) * 192 + d] = KPE[(size_t)t * DROPE + (d - DNOPE)];
    } else {
        int dv = d - 192;
        Vt[(size_t)h * DVAL * T_TOK + (size_t)dv * T_TOK + t] =
            (__bf16)KVB[(size_t)t * (NHEAD * 256) + (size_t)h * 256 + DNOPE + dv];
    }
}

// ---------------------------------------------------------------------------
// Flash attention, block-cooperative:
// 4 waves/block, 64 query rows/block (16 per wave), one head per blockIdx.y.
// Each 32-col s-tile: K tile (32x192) and V^T tile (128x32) staged ONCE in
// LDS by all 128 threads, then consumed by all 4 waves as WMMA B fragments
// via ds_load_b128 -> 4x less K/V global traffic. Online softmax uses 16-lane
// xor-shuffle row reductions (C-layout: a row's 16 cols sit in one acc reg
// across a half-wave). P staged in per-wave LDS, re-read as A fragment.
// ---------------------------------------------------------------------------
__global__ __launch_bounds__(128) void mla_attention(const __bf16* __restrict__ Qb,
                                                     const __bf16* __restrict__ Kb,
                                                     const __bf16* __restrict__ Vt,
                                                     __bf16* __restrict__ Obf) {
    __shared__ __bf16 Ktile[32][192];   // 12 KB
    __shared__ __bf16 Vtile[128][32];   //  8 KB
    __shared__ __bf16 plds[4][16][32];  //  4 KB
    const int tid  = threadIdx.x;
    const int wave = tid >> 5;
    const int lane = tid & 31;
    const int h       = blockIdx.y;
    const int qt_blk  = blockIdx.x * 64;
    const int qt      = qt_blk + wave * 16;

    const __bf16* Qh = Qb + (size_t)h * T_TOK * 192;
    const __bf16* Kh = Kb + (size_t)h * T_TOK * 192;
    const __bf16* Vh = Vt + (size_t)h * DVAL * T_TOK;

    v16bf qf[6];
#pragma unroll
    for (int kk = 0; kk < 6; ++kk) qf[kk] = load_frag16(Qh, 192, qt, kk * 32);

    float m8[8], l8[8];
    v8f o[8];
#pragma unroll
    for (int r = 0; r < 8; ++r) { m8[r] = -3.0e38f; l8[r] = 0.f; }
#pragma unroll
    for (int j = 0; j < 8; ++j) o[j] = 0.f;

    const int ro = (lane >> 4) << 3;   // row offset of this lane's C rows
    const int co = lane & 15;          // column within 16-wide n-subtile

    const int smax = qt_blk + 64;      // uniform per block
    for (int s0 = 0; s0 < smax; s0 += 32) {
        // ---- cooperative K/V tile staging (128 threads) ----
        __syncthreads();  // previous tile fully consumed
#pragma unroll
        for (int c = 0; c < 6; ++c) {          // 32*192/8 = 768 v8bf chunks
            int ch  = tid + 128 * c;
            int row = ch / 24, col = (ch % 24) * 8;
            *(v8bf*)&Ktile[row][col] =
                *(const v8bf*)&Kh[(size_t)(s0 + row) * 192 + col];
        }
#pragma unroll
        for (int c = 0; c < 4; ++c) {          // 128*32/8 = 512 v8bf chunks
            int ch  = tid + 128 * c;
            int row = ch / 4, col = (ch % 4) * 8;
            *(v8bf*)&Vtile[row][col] =
                *(const v8bf*)&Vh[(size_t)row * T_TOK + s0 + col];
        }
        __syncthreads();  // tile visible to all waves

        if (s0 <= qt) {   // skip tiles entirely above this wave's diagonal
            v8f s[2];
            s[0] = 0.f; s[1] = 0.f;
#pragma unroll
            for (int kk = 0; kk < 6; ++kk) {
                v16bf b0 = load_frag16(&Ktile[0][0], 192, 0,  kk * 32);
                v16bf b1 = load_frag16(&Ktile[0][0], 192, 16, kk * 32);
                s[0] = __builtin_amdgcn_wmma_f32_16x16x32_bf16(
                    false, qf[kk], false, b0, (short)0, s[0], false, false);
                s[1] = __builtin_amdgcn_wmma_f32_16x16x32_bf16(
                    false, qf[kk], false, b1, (short)0, s[1], false, false);
            }

#pragma unroll
            for (int r = 0; r < 8; ++r) {
                const int row = qt + r + ro;
                float v0 = s[0][r] * ATT_SCALE;
                float v1 = s[1][r] * ATT_SCALE;
                if (s0 + co      > row) v0 = -3.0e38f;
                if (s0 + 16 + co > row) v1 = -3.0e38f;

                float mx = fmaxf(v0, v1);
                mx = fmaxf(mx, __shfl_xor(mx, 1));
                mx = fmaxf(mx, __shfl_xor(mx, 2));
                mx = fmaxf(mx, __shfl_xor(mx, 4));
                mx = fmaxf(mx, __shfl_xor(mx, 8));

                const float mnew  = fmaxf(m8[r], mx);
                const float alpha = __expf(m8[r] - mnew);
                const float p0 = __expf(v0 - mnew);
                const float p1 = __expf(v1 - mnew);
                float rs = p0 + p1;
                rs += __shfl_xor(rs, 1);
                rs += __shfl_xor(rs, 2);
                rs += __shfl_xor(rs, 4);
                rs += __shfl_xor(rs, 8);
                l8[r] = l8[r] * alpha + rs;
                m8[r] = mnew;

                plds[wave][r + ro][co]      = (__bf16)p0;
                plds[wave][r + ro][16 + co] = (__bf16)p1;
#pragma unroll
                for (int j = 0; j < 8; ++j) o[j][r] *= alpha;
            }
            asm volatile("s_wait_dscnt 0" ::: "memory");

            v16bf pf = load_frag16(&plds[wave][0][0], 32, 0, 0);
#pragma unroll
            for (int j = 0; j < 8; ++j) {
                v16bf bv = load_frag16(&Vtile[0][0], 32, 16 * j, 0);
                o[j] = __builtin_amdgcn_wmma_f32_16x16x32_bf16(
                    false, pf, false, bv, (short)0, o[j], false, false);
            }
        }
    }

#pragma unroll
    for (int j = 0; j < 8; ++j)
#pragma unroll
        for (int r = 0; r < 8; ++r) {
            const int trow = qt + r + ro;
            Obf[(size_t)trow * (NHEAD * DVAL) + h * DVAL + 16 * j + co] =
                (__bf16)(o[j][r] / l8[r]);
        }
}

// ---------------------------------------------------------------------------
// Orchestration
// ---------------------------------------------------------------------------
extern "C" void kernel_launch(void* const* d_in, const int* in_sizes, int n_in,
                              void* d_out, int out_size, void* d_ws, size_t ws_size,
                              hipStream_t stream) {
    (void)in_sizes; (void)n_in; (void)out_size; (void)ws_size;
    const float* hidden  = (const float*)d_in[1];
    const float* wq_a    = (const float*)d_in[2];
    const float* q_a_ln  = (const float*)d_in[3];
    const float* wq_b    = (const float*)d_in[4];
    const float* wkv_a   = (const float*)d_in[5];
    const float* kv_a_ln = (const float*)d_in[6];
    const float* wkv_b   = (const float*)d_in[7];
    const float* wo      = (const float*)d_in[8];
    float* out = (float*)d_out;

    char* ws = (char*)d_ws;
    size_t off = 0;
    auto alloc = [&](size_t bytes) -> void* {
        void* p = ws + off;
        off = (off + bytes + 255) & ~(size_t)255;
        return p;
    };

    __bf16* hidden_bf = (__bf16*)alloc((size_t)T_TOK * HID * 2);
    __bf16* wqaT  = (__bf16*)alloc((size_t)QLAT * HID * 2);            // [1536,5120]
    __bf16* wqbT  = (__bf16*)alloc((size_t)3072 * QLAT * 2);           // [3072,1536]
    __bf16* wkvaT = (__bf16*)alloc((size_t)640 * HID * 2);             // [640,5120] padded
    __bf16* wkvbT = (__bf16*)alloc((size_t)4096 * KVLAT * 2);          // [4096,512]
    __bf16* woT   = (__bf16*)alloc((size_t)HID * 2048 * 2);            // [5120,2048]
    float*  qlat  = (float*)alloc((size_t)T_TOK * QLAT * 4);           // [2048,1536]
    __bf16* qan   = (__bf16*)alloc((size_t)T_TOK * QLAT * 2);
    float*  qf32  = (float*)alloc((size_t)T_TOK * 3072 * 4);           // [2048,16,192]
    float*  kvf32 = (float*)alloc((size_t)T_TOK * 640 * 4);            // ld 640
    __bf16* ckv   = (__bf16*)alloc((size_t)T_TOK * KVLAT * 2);
    __bf16* kpe   = (__bf16*)alloc((size_t)T_TOK * DROPE * 2);
    float*  kvbf  = (float*)alloc((size_t)T_TOK * 4096 * 4);           // [2048,16,256]
    __bf16* Qb    = (__bf16*)alloc((size_t)NHEAD * T_TOK * 192 * 2);
    __bf16* Kb    = (__bf16*)alloc((size_t)NHEAD * T_TOK * 192 * 2);
    __bf16* Vt    = (__bf16*)alloc((size_t)NHEAD * DVAL * T_TOK * 2);
    __bf16* attnb = (__bf16*)alloc((size_t)T_TOK * 2048 * 2);

    // --- convert inputs to bf16 (weights transposed to [N,K]) ---
    {
        size_t n = (size_t)T_TOK * HID;
        conv_bf16<<<dim3((unsigned)((n + 255) / 256)), 256, 0, stream>>>(hidden, hidden_bf, n);
    }
    transpose_conv<<<dim3(QLAT / 32, HID / 32),  256, 0, stream>>>(wq_a,  wqaT,  HID,  QLAT, QLAT);
    transpose_conv<<<dim3(3072 / 32, QLAT / 32), 256, 0, stream>>>(wq_b,  wqbT,  QLAT, 3072, 3072);
    transpose_conv<<<dim3(640 / 32,  HID / 32),  256, 0, stream>>>(wkv_a, wkvaT, HID,  576,  640);
    transpose_conv<<<dim3(4096 / 32, KVLAT / 32),256, 0, stream>>>(wkv_b, wkvbT, KVLAT,4096, 4096);
    transpose_conv<<<dim3(HID / 32,  2048 / 32), 256, 0, stream>>>(wo,    woT,   2048, HID,  HID);

    // --- q path: hidden @ wq_a -> rmsnorm -> @ wq_b ---
    gemm_bf16_wmma<<<dim3(QLAT / 128, T_TOK / 128), 128, 0, stream>>>(
        hidden_bf, wqaT, qlat, T_TOK, QLAT, HID);
    rmsnorm_bf16<<<T_TOK, 256, 0, stream>>>(qlat, q_a_ln, qan, QLAT);
    gemm_bf16_wmma<<<dim3(3072 / 128, T_TOK / 128), 128, 0, stream>>>(
        qan, wqbT, qf32, T_TOK, 3072, QLAT);

    // --- kv path: hidden @ wkv_a (N padded to 640) -> norm/rope -> @ wkv_b ---
    gemm_bf16_wmma<<<dim3(640 / 128, T_TOK / 128), 128, 0, stream>>>(
        hidden_bf, wkvaT, kvf32, T_TOK, 640, HID);
    prep_kv<<<T_TOK, 256, 0, stream>>>(kvf32, kv_a_ln, ckv, kpe);
    gemm_bf16_wmma<<<dim3(4096 / 128, T_TOK / 128), 128, 0, stream>>>(
        ckv, wkvbT, kvbf, T_TOK, 4096, KVLAT);

    // --- attention operand layouts ---
    {
        size_t n = (size_t)T_TOK * NHEAD * 96;
        build_q<<<dim3((unsigned)((n + 255) / 256)), 256, 0, stream>>>(qf32, Qb);
    }
    {
        size_t n = (size_t)NHEAD * T_TOK * 320;
        build_kv<<<dim3((unsigned)((n + 255) / 256)), 256, 0, stream>>>(kvbf, kpe, Kb, Vt);
    }

    // --- flash attention (64 query rows per block, LDS-staged K/V) ---
    mla_attention<<<dim3(T_TOK / 64, NHEAD), 128, 0, stream>>>(Qb, Kb, Vt, attnb);

    // --- output projection ---
    gemm_bf16_wmma<<<dim3(HID / 128, T_TOK / 128), 128, 0, stream>>>(
        attnb, woT, out, T_TOK, HID, 2048);
}
